// SAM_71854802862489
// MI455X (gfx1250) — matrix operand order repaired and checked by reading, not compiled, not run on hardware
//
#include <hip/hip_runtime.h>
#include <hip/hip_bf16.h>

typedef __attribute__((ext_vector_type(16))) __bf16 v16bf;
typedef __attribute__((ext_vector_type(8)))  float  v8f;
typedef unsigned short u16;

// ---------------------------------------------------------------------------
// helpers
// ---------------------------------------------------------------------------
__device__ __forceinline__ u16 f2bf(float f) {
  unsigned u = __float_as_uint(f);
  u += 0x7FFFu + ((u >> 16) & 1u);   // round-to-nearest-even
  return (u16)(u >> 16);
}

union FragU { v16bf v; uint4 q[2]; };

// A-operand (16x32 bf16): lane r=l%16 holds row r; half h=l/16 holds
// K-chunks [8h..8h+7] and [16+8h..16+8h+7]  (two contiguous 16B loads)
__device__ __forceinline__ v16bf load_frag_a(const u16* __restrict__ p, int ld,
                                             int row0, int k0) {
  const int lane = threadIdx.x & 31;
  const int r = lane & 15, h = lane >> 4;
  const u16* q = p + (size_t)(row0 + r) * ld + k0 + 8 * h;
  FragU f;
  f.q[0] = *reinterpret_cast<const uint4*>(q);
  f.q[1] = *reinterpret_cast<const uint4*>(q + 16);
  return f.v;
}

// B-operand (32x16 bf16): lane r=l%16 holds column n0+r; half h=l/16 holds
// K-chunks [8h..8h+7] and [16+8h..16+8h+7] -> source must be N-major (B^T rows)
__device__ __forceinline__ v16bf load_frag_b(const u16* __restrict__ p, int ld,
                                             int n0, int k0) {
  const int lane = threadIdx.x & 31;
  const int r = lane & 15, h = lane >> 4;
  const u16* q = p + (size_t)(n0 + r) * ld + k0 + 8 * h;
  FragU f;
  f.q[0] = *reinterpret_cast<const uint4*>(q);
  f.q[1] = *reinterpret_cast<const uint4*>(q + 16);
  return f.v;
}

// ---------------------------------------------------------------------------
// prep kernels
// ---------------------------------------------------------------------------
__global__ void prep_w(const float* __restrict__ W1, const float* __restrict__ W2,
                       u16* __restrict__ W1T, u16* __restrict__ W2T) {
  int idx = blockIdx.x * blockDim.x + threadIdx.x;   // 131072 = 2*256*256
  int wsel = idx >> 16;
  int rem  = idx & 65535;
  int k = rem >> 8, n = rem & 255;
  const float* W = wsel ? W2 : W1;
  u16* WT = wsel ? W2T : W1T;
  WT[n * 256 + k] = f2bf(W[k * 256 + n]);
}

__global__ void prep_x(const float* __restrict__ x, u16* __restrict__ xb) {
  int idx = blockIdx.x * blockDim.x + threadIdx.x;   // 4194304 elems
  xb[idx] = f2bf(x[idx]);
}

// xT[b][c][n] = bf16(x[b][n][c]) ; 32x32 LDS tile transpose
__global__ void prep_xT(const float* __restrict__ x, u16* __restrict__ xT) {
  __shared__ float t[32][33];
  const int b = blockIdx.z;
  const int c0 = blockIdx.x * 32, n0 = blockIdx.y * 32;
  const int tx = threadIdx.x, ty = threadIdx.y;   // (32, 8)
  #pragma unroll
  for (int i = 0; i < 4; ++i)
    t[ty + 8 * i][tx] = x[((size_t)b * 1024 + n0 + ty + 8 * i) * 256 + c0 + tx];
  __syncthreads();
  #pragma unroll
  for (int i = 0; i < 4; ++i)
    xT[((size_t)b * 256 + c0 + ty + 8 * i) * 1024 + n0 + tx] = f2bf(t[tx][ty + 8 * i]);
}

// ---------------------------------------------------------------------------
// K1: A = x@W1+b1 and Bp = x@W2+b2 (bf16 out). grid (1024, 1, 2), block 128.
// Each wave: 16m x 64n (4 accumulators), A-fragment reused across 4 n-tiles.
// ---------------------------------------------------------------------------
__global__ __launch_bounds__(128) void k1_gemm(const u16* __restrict__ xb,
    const u16* __restrict__ W1T, const u16* __restrict__ W2T,
    const float* __restrict__ b1, const float* __restrict__ b2,
    u16* __restrict__ Abuf, u16* __restrict__ Bbuf) {
  const int z = blockIdx.z;
  const u16*   WT   = z ? W2T : W1T;
  const float* bias = z ? b2 : b1;
  u16* outp = z ? Bbuf : Abuf;
  const int m0 = blockIdx.x * 16;
  const int w  = threadIdx.x >> 5;
  const int nbase = w * 64;
  v8f c[4] = {{}, {}, {}, {}};
  #pragma unroll
  for (int kt = 0; kt < 8; ++kt) {
    v16bf a = load_frag_a(xb, 256, m0, kt * 32);
    #pragma unroll
    for (int t = 0; t < 4; ++t) {
      v16bf bb = load_frag_b(WT, 256, nbase + 16 * t, kt * 32);
      c[t] = __builtin_amdgcn_wmma_f32_16x16x32_bf16(false, a, false, bb, (short)0,
                                                     c[t], false, false);
    }
  }
  const int lane = threadIdx.x & 31;
  const int r = lane & 15, h = lane >> 4;
  #pragma unroll
  for (int t = 0; t < 4; ++t) {
    const int col = nbase + 16 * t + r;
    const float bv = bias[col];
    #pragma unroll
    for (int v = 0; v < 8; ++v) {
      int row = m0 + v + 8 * h;
      outp[(size_t)row * 256 + col] = f2bf(c[t][v] + bv);
    }
  }
}

// ---------------------------------------------------------------------------
// K2: S = A@Bp^T row-strip, row softmax, write F^T bf16.
// grid (64, 16), block 32 (one wave), 64KB LDS. A-fragments live in VGPRs.
// ---------------------------------------------------------------------------
__global__ __launch_bounds__(32) void k2_scores(const u16* __restrict__ A,
                                                const u16* __restrict__ Bp,
                                                u16* __restrict__ Ft) {
  __shared__ float S[16 * 1024];                 // XOR-swizzled: [r][j^r]
  const int b  = blockIdx.y;
  const int i0 = blockIdx.x * 16;
  const u16* Ab = A  + (size_t)b * 1024 * 256;
  const u16* Bb = Bp + (size_t)b * 1024 * 256;
  const int lane = threadIdx.x & 31;
  const int r = lane & 15, h = lane >> 4;

  v16bf a[8];
  #pragma unroll
  for (int kt = 0; kt < 8; ++kt) a[kt] = load_frag_a(Ab, 256, i0, kt * 32);

  for (int jt = 0; jt < 64; ++jt) {
    v8f c = {};
    #pragma unroll
    for (int kt = 0; kt < 8; ++kt) {
      v16bf bb = load_frag_b(Bb, 256, jt * 16, kt * 32);
      c = __builtin_amdgcn_wmma_f32_16x16x32_bf16(false, a[kt], false, bb, (short)0,
                                                  c, false, false);
    }
    #pragma unroll
    for (int v = 0; v < 8; ++v) {
      int row = v + 8 * h;
      int col = jt * 16 + r;
      S[row * 1024 + (col ^ row)] = c[v];
    }
  }
  __syncthreads();

  // row softmax: lane handles row r over half h of the 1024 columns
  float m = -3.0e38f;
  for (int j = h * 512; j < h * 512 + 512; ++j)
    m = fmaxf(m, S[r * 1024 + (j ^ r)]);
  m = fmaxf(m, __shfl_xor(m, 16, 32));
  float s = 0.f;
  for (int j = h * 512; j < h * 512 + 512; ++j)
    s += __expf(S[r * 1024 + (j ^ r)] - m);
  s += __shfl_xor(s, 16, 32);
  const float inv = 1.0f / s;

  // write transposed: Ft[b][j][i0+r]  (16 consecutive bf16 per half-wave per j)
  u16* F0 = Ft + (size_t)b * 1024 * 1024 + i0 + r;
  for (int jj = 0; jj < 512; ++jj) {
    int j = jj * 2 + h;
    float p = __expf(S[r * 1024 + (j ^ r)] - m) * inv;
    F0[(size_t)j * 1024] = f2bf(p);
  }
}

// ---------------------------------------------------------------------------
// K3: M = F^T @ x (K=1024), out = x + kappa*M.  grid (32, 16), block 128.
// Each wave: 32j x 64c block (2x4 accumulators); block covers 32j x 256c.
// Per k-step: 2 A-frags + 4 B-frags feed 8 WMMAs.
// ---------------------------------------------------------------------------
__global__ __launch_bounds__(128) void k3_out(const u16* __restrict__ Ft,
                                              const u16* __restrict__ xT,
                                              const float* __restrict__ x,
                                              const float* __restrict__ kappa,
                                              float* __restrict__ out) {
  const int b  = blockIdx.y;
  const int j0 = blockIdx.x * 32;
  const int w  = threadIdx.x >> 5;
  const int cbase = w * 64;
  const u16* Fb = Ft + (size_t)b * 1024 * 1024;
  const u16* Xb = xT + (size_t)b * 256 * 1024;
  v8f acc[8] = {{}, {}, {}, {}, {}, {}, {}, {}};
  for (int kt = 0; kt < 32; ++kt) {
    v16bf a0 = load_frag_a(Fb, 1024, j0,      kt * 32);
    v16bf a1 = load_frag_a(Fb, 1024, j0 + 16, kt * 32);
    #pragma unroll
    for (int t = 0; t < 4; ++t) {
      v16bf bb = load_frag_b(Xb, 1024, cbase + 16 * t, kt * 32);
      acc[t]     = __builtin_amdgcn_wmma_f32_16x16x32_bf16(false, a0, false, bb,
                                                           (short)0, acc[t], false, false);
      acc[4 + t] = __builtin_amdgcn_wmma_f32_16x16x32_bf16(false, a1, false, bb,
                                                           (short)0, acc[4 + t], false, false);
    }
  }
  const float kap = kappa[0];
  const int lane = threadIdx.x & 31;
  const int r = lane & 15, h = lane >> 4;
  #pragma unroll
  for (int jj = 0; jj < 2; ++jj) {
    #pragma unroll
    for (int t = 0; t < 4; ++t) {
      #pragma unroll
      for (int v = 0; v < 8; ++v) {
        int row = j0 + 16 * jj + v + 8 * h;
        size_t idx = ((size_t)b * 1024 + row) * 256 + cbase + 16 * t + r;
        out[idx] = x[idx] + kap * acc[4 * jj + t][v];
      }
    }
  }
}

// ---------------------------------------------------------------------------
// launch
// ---------------------------------------------------------------------------
extern "C" void kernel_launch(void* const* d_in, const int* in_sizes, int n_in,
                              void* d_out, int out_size, void* d_ws, size_t ws_size,
                              hipStream_t stream) {
  const float* x     = (const float*)d_in[0];
  const float* W1    = (const float*)d_in[1];
  const float* b1    = (const float*)d_in[2];
  const float* W2    = (const float*)d_in[3];
  const float* b2    = (const float*)d_in[4];
  const float* kappa = (const float*)d_in[5];
  float* out = (float*)d_out;

  char* ws = (char*)d_ws;
  u16* xb   = (u16*)(ws + 0);          //  8 MiB  x bf16 row-major [16384,256]
  u16* xT   = (u16*)(ws + 8388608);    //  8 MiB  x^T bf16 [16,256,1024]
  u16* W1T  = (u16*)(ws + 16777216);   // 128 KiB
  u16* W2T  = (u16*)(ws + 16908288);   // 128 KiB
  u16* Abuf = (u16*)(ws + 17039360);   //  8 MiB  A bf16
  u16* Bbuf = (u16*)(ws + 25427968);   //  8 MiB  Bp bf16
  u16* Ft   = (u16*)(ws + 33816576);   // 32 MiB  F^T bf16 [16,1024,1024]

  prep_w  <<<dim3(512),        dim3(256),   0, stream>>>(W1, W2, W1T, W2T);
  prep_x  <<<dim3(16384),      dim3(256),   0, stream>>>(x, xb);
  prep_xT <<<dim3(8, 32, 16),  dim3(32, 8), 0, stream>>>(x, xT);
  k1_gemm <<<dim3(1024, 1, 2), dim3(128),   0, stream>>>(xb, W1T, W2T, b1, b2,
                                                         Abuf, Bbuf);
  k2_scores<<<dim3(64, 16),    dim3(32),    0, stream>>>(Abuf, Bbuf, Ft);
  k3_out  <<<dim3(32, 16),     dim3(128),   0, stream>>>(Ft, xT, x, kappa, out);
}
